// QEffDeepseekV3MoE_34643206210149
// MI455X (gfx1250) — matrix operand order repaired and checked by reading, not compiled, not run on hardware
//
#include <hip/hip_runtime.h>
#include <stdint.h>

// ---------------- problem constants ----------------
#define T_TOK   1024
#define H_DIM   2048
#define I_DIM   1408
#define E_NUM   32
#define TOPK    4
#define GS_Q    128
#define ISH_DIM 2816
#define ROUTED_SCALE 2.5f
#define MAX_TILES 288               // 4096/16 routed rows + up to 32 pad tiles
#define MAX_ROWS  (MAX_TILES * 16)  // 4608

typedef __attribute__((ext_vector_type(16))) _Float16 v16h;
typedef __attribute__((ext_vector_type(8)))  float    v8f;

// ---------------- helpers ----------------
__device__ __forceinline__ uint32_t pk2h(float a, float b) {
  union { _Float16 h[2]; uint32_t u; } r;
  r.h[0] = (_Float16)a; r.h[1] = (_Float16)b;
  return r.u;
}

// fast silu: g * v_rcp_f32(1 + exp(-g))  (no IEEE div_scale chains)
__device__ __forceinline__ float silu_f(float g) {
  return g * __builtin_amdgcn_rcpf(1.f + __expf(-g));
}

// A fragment: M x K(sub)=16x32 f16 slice of a [16][64] LDS tile (stride 64).
// ISA layout: lanes (kgrp*16+m); VGPR r: K = (r<4?0:16) + kgrp*8 + 2*(r&3)
__device__ __forceinline__ v16h load_a_frag(const _Float16* lds_a, int lane, int koff) {
  union { v16h v; uint32_t u[8]; } a;
  const int m  = lane & 15;
  const int kb = (lane >> 4) * 8;
  const uint32_t* row = (const uint32_t*)(lds_a + m * 64 + koff);
#pragma unroll
  for (int r = 0; r < 8; ++r) {
    int k = ((r < 4) ? 0 : 16) + kb + 2 * (r & 3);
    a.u[r] = row[k >> 1];
  }
  return a.v;
}

// B fragment: 32x16 slice of B^T stored in LDS as [16 n][64 k] (stride 64).
// ISA layout: lane = kgrp*16+n; VGPR r: K = (r<4?0:16) + kgrp*8 + 2*(r&3)
__device__ __forceinline__ v16h load_b_frag(const _Float16* lds_b, int lane, int koff) {
  union { v16h v; uint32_t u[8]; } b;
  const int n  = lane & 15;
  const int kb = (lane >> 4) * 8;
  const uint32_t* row = (const uint32_t*)(lds_b + n * 64 + koff + kb);
#pragma unroll
  for (int r = 0; r < 8; ++r) b.u[r] = row[(r < 4) ? r : r + 4];
  return b.v;
}

__device__ __forceinline__ v8f wmma_f16(v16h a, v16h b, v8f c) {
  return __builtin_amdgcn_wmma_f32_16x16x32_f16(false, a, false, b, (short)0, c,
                                                false, false);
}

// GPTQ dequant of a [16 out][64 in] tile into LDS (B^T layout, stride 64).
// One wave: lane -> (oloc = lane>>1, half = lane&1), 4 int32 words each.
// k0 is a multiple of 64 so the tile sits inside one GS=128 group.
__device__ __forceinline__ void dequant_tile64(
    const int* __restrict__ qw, const int* __restrict__ qz,
    const float* __restrict__ qs, int ldWords, int ldGroups,
    int outBase, int k0, _Float16* lds, int lane)
{
  const int oloc = lane >> 1;
  const int wsel = lane & 1;
  const int o = outBase + oloc;
  const int g = k0 >> 7;
  const float s = qs[o * ldGroups + g];
  const float nzs = -(float)qz[o * ldGroups + g] * s;   // (c - z)*s = c*s + nzs
  const int wbase = (k0 >> 3) + wsel * 4;
  const int* wsrc = qw + (size_t)o * ldWords + wbase;
  __builtin_prefetch(wsrc + 8, 0, 0);                   // next K-tile's words
  uint32_t* dst = (uint32_t*)(lds + oloc * 64 + wsel * 32);
#pragma unroll
  for (int wi = 0; wi < 4; ++wi) {
    uint32_t q = (uint32_t)wsrc[wi];
#pragma unroll
    for (int j = 0; j < 4; ++j) {
      float w0 = fmaf((float)((q >> (8 * j))     & 0xF), s, nzs);
      float w1 = fmaf((float)((q >> (8 * j + 4)) & 0xF), s, nzs);
      dst[wi * 4 + j] = pk2h(w0, w1);
    }
  }
}

// ---------------- kernel 1: zero counts ----------------
__global__ void zero_counts_kernel(int* counts) {
  if (threadIdx.x < E_NUM) counts[threadIdx.x] = 0;
}

// ---------------- kernel 2: router (1 wave32 per token) ----------------
__global__ __launch_bounds__(32) void router_kernel(
    const float* __restrict__ x, const float* __restrict__ gw,
    int* __restrict__ tokexp, float* __restrict__ tokw, int* __restrict__ counts)
{
  const int t = blockIdx.x;
  const int e = threadIdx.x;                 // 0..31 == expert id
  const float* xr = x + (size_t)t * H_DIM;
  float acc = 0.f;
  for (int k = 0; k < H_DIM; ++k) acc += xr[k] * gw[k * E_NUM + e];
  float s = __builtin_amdgcn_rcpf(1.f + __expf(-acc));   // sigmoid

  float vals[TOPK]; int sel[TOPK]; float ssum = 0.f;
#pragma unroll
  for (int it = 0; it < TOPK; ++it) {
    float m = s; int mi = e;
    for (int off = 16; off; off >>= 1) {
      float om = __shfl_xor(m, off);
      int   oi = __shfl_xor(mi, off);
      if (om > m || (om == m && oi < mi)) { m = om; mi = oi; }
    }
    vals[it] = m; sel[it] = mi; ssum += m;
    if (e == mi) s = -1e30f;
  }
  if (e < TOPK) {
    float w = vals[e] * __builtin_amdgcn_rcpf(ssum + 1e-20f) * ROUTED_SCALE;
    tokexp[t * TOPK + e] = sel[e];
    tokw[t * TOPK + e] = w;
    atomicAdd(&counts[sel[e]], 1);
  }
}

// ---------------- kernel 3: scheduler (16-aligned prefix sum) ----------------
__global__ __launch_bounds__(256) void sched_kernel(
    const int* __restrict__ counts, int* __restrict__ offsets,
    int* __restrict__ cursor, int* __restrict__ tileExpert,
    int* __restrict__ rowmap, float* __restrict__ roww)
{
  __shared__ int soff[E_NUM + 1];
  if (threadIdx.x == 0) {
    int off = 0;
    for (int ee = 0; ee < E_NUM; ++ee) {
      soff[ee] = off;
      off += (counts[ee] + 15) & ~15;        // pad each expert to tile multiple
    }
    soff[E_NUM] = off;
  }
  __syncthreads();
  if (threadIdx.x < E_NUM) {
    offsets[threadIdx.x] = soff[threadIdx.x];
    cursor[threadIdx.x] = soff[threadIdx.x];
  }
  if (threadIdx.x == 0) offsets[E_NUM] = soff[E_NUM];
  const int nTiles = soff[E_NUM] >> 4;
  for (int tl = threadIdx.x; tl < MAX_TILES; tl += blockDim.x) {
    int ex = -1;
    if (tl < nTiles) {
      int r = tl * 16;
      for (int ee = 0; ee < E_NUM; ++ee)
        if (r >= soff[ee] && r < soff[ee + 1]) ex = ee;
    }
    tileExpert[tl] = ex;
  }
  for (int r = threadIdx.x; r < MAX_ROWS; r += blockDim.x) {
    rowmap[r] = -1;
    roww[r] = 0.f;
  }
}

// ---------------- kernel 4: scatter tokens into expert row lists -------------
__global__ void fill_rows_kernel(
    const int* __restrict__ tokexp, const float* __restrict__ tokw,
    int* __restrict__ cursor, int* __restrict__ rowmap, float* __restrict__ roww)
{
  int idx = blockIdx.x * blockDim.x + threadIdx.x;
  if (idx >= T_TOK * TOPK) return;
  int e = tokexp[idx];
  int pos = atomicAdd(&cursor[e], 1);
  rowmap[pos] = idx >> 2;
  roww[pos] = tokw[idx];
}

// ---------------- kernel 5: x -> fp16 ----------------
__global__ void cvt_x_kernel(const float* __restrict__ x, _Float16* __restrict__ xh) {
  int i = blockIdx.x * blockDim.x + threadIdx.x;
  if (i < T_TOK * H_DIM) xh[i] = (_Float16)x[i];
}

// Cooperative [16][64] f16 A-tile load: 512 dwords over 256 threads.
__device__ __forceinline__ void load_a_tile(
    _Float16* lds_a, const _Float16* __restrict__ src, size_t rowStride, int kk, int tid)
{
#pragma unroll
  for (int rep = 0; rep < 2; ++rep) {
    int dw = rep * 256 + tid;
    int m = dw >> 5;
    int kp = (dw & 31) * 2;
    *(uint32_t*)(lds_a + m * 64 + kp) =
        *(const uint32_t*)(src + (size_t)m * rowStride + kk + kp);
  }
}

// Per-wave B^T tile load with fp32->fp16 convert + packed b32 LDS stores.
// W is [K, N] row-major; tile covers rows kk..kk+63, cols n0..n0+15.
__device__ __forceinline__ void load_bT_tile(
    _Float16* lds_b, const float* __restrict__ W, int ldN, int kk, int n0, int lane)
{
#pragma unroll
  for (int p = 0; p < 16; ++p) {
    int pairIdx = p * 32 + lane;            // 0..511
    int n = pairIdx & 15;
    int kp = (pairIdx >> 4) * 2;            // even k
    float w0 = W[(size_t)(kk + kp)     * ldN + n0 + n];
    float w1 = W[(size_t)(kk + kp + 1) * ldN + n0 + n];
    *(uint32_t*)(lds_b + n * 64 + kp) = pk2h(w0, w1);
  }
}

// ---------------- kernel 6: shared experts gate/up (dense) -------------------
__global__ __launch_bounds__(256) void shared_gateup_kernel(
    const _Float16* __restrict__ xh, const float* __restrict__ sgw,
    const float* __restrict__ suw, _Float16* __restrict__ hsbuf)
{
  __shared__ _Float16 lds_a[16 * 64];
  __shared__ _Float16 lds_bg[8][16 * 64];
  __shared__ _Float16 lds_bu[8][16 * 64];
  const int wave = threadIdx.x >> 5, lane = threadIdx.x & 31;
  const int t0 = blockIdx.x * 16;
  const int n0 = blockIdx.y * 128 + wave * 16;
  v8f accg = {}; v8f accu = {};
  for (int kk = 0; kk < H_DIM; kk += 64) {
    load_a_tile(lds_a, xh + (size_t)t0 * H_DIM, H_DIM, kk, threadIdx.x);
    load_bT_tile(lds_bg[wave], sgw, ISH_DIM, kk, n0, lane);
    load_bT_tile(lds_bu[wave], suw, ISH_DIM, kk, n0, lane);
    __syncthreads();
#pragma unroll
    for (int ko = 0; ko < 64; ko += 32) {
      v16h a = load_a_frag(lds_a, lane, ko);
      accg = wmma_f16(a, load_b_frag(lds_bg[wave], lane, ko), accg);
      accu = wmma_f16(a, load_b_frag(lds_bu[wave], lane, ko), accu);
    }
    __syncthreads();
  }
  const int n = lane & 15, mb = (lane >> 4) * 8;
#pragma unroll
  for (int r = 0; r < 8; ++r) {
    float hv = silu_f(accg[r]) * accu[r];
    hsbuf[(size_t)(t0 + mb + r) * ISH_DIM + n0 + n] = (_Float16)hv;
  }
}

// ---------------- kernel 7: shared experts down-proj (assigns out) -----------
__global__ __launch_bounds__(256) void shared_down_kernel(
    const _Float16* __restrict__ hsbuf, const float* __restrict__ sdw,
    float* __restrict__ out)
{
  __shared__ _Float16 lds_a[16 * 64];
  __shared__ _Float16 lds_b[8][16 * 64];
  const int wave = threadIdx.x >> 5, lane = threadIdx.x & 31;
  const int t0 = blockIdx.x * 16;
  const int n0 = blockIdx.y * 128 + wave * 16;
  v8f acc = {};
  for (int kk = 0; kk < ISH_DIM; kk += 64) {
    load_a_tile(lds_a, hsbuf + (size_t)t0 * ISH_DIM, ISH_DIM, kk, threadIdx.x);
    load_bT_tile(lds_b[wave], sdw, H_DIM, kk, n0, lane);
    __syncthreads();
#pragma unroll
    for (int ko = 0; ko < 64; ko += 32)
      acc = wmma_f16(load_a_frag(lds_a, lane, ko),
                     load_b_frag(lds_b[wave], lane, ko), acc);
    __syncthreads();
  }
  const int n = lane & 15, mb = (lane >> 4) * 8;
#pragma unroll
  for (int r = 0; r < 8; ++r)
    out[(size_t)(t0 + mb + r) * H_DIM + n0 + n] = acc[r];   // assign (init)
}

// ---------------- kernel 8: routed experts gate/up + SiLU fuse ---------------
__global__ __launch_bounds__(256) void moe_gateup_kernel(
    const _Float16* __restrict__ xh, const int* __restrict__ rowmap,
    const int* __restrict__ tileExpert,
    const int* __restrict__ gq, const int* __restrict__ gz, const float* __restrict__ gsc,
    const int* __restrict__ uq, const int* __restrict__ uz, const float* __restrict__ usc,
    _Float16* __restrict__ hbuf)
{
  __shared__ _Float16 lds_a[16 * 64];
  __shared__ _Float16 lds_bg[8][16 * 64];
  __shared__ _Float16 lds_bu[8][16 * 64];
  __shared__ int toks[16];

  const int tile = blockIdx.x;
  const int e = tileExpert[tile];
  if (e < 0) return;
  const int wave = threadIdx.x >> 5, lane = threadIdx.x & 31;
  const int r0 = tile * 16;
  const int iCol0 = blockIdx.y * 128 + wave * 16;
  const int ldW = H_DIM / 8, ldG = H_DIM / GS_Q;
  const int*   gqe = gq  + (size_t)e * I_DIM * ldW;
  const int*   gze = gz  + (size_t)e * I_DIM * ldG;
  const float* gse = gsc + (size_t)e * I_DIM * ldG;
  const int*   uqe = uq  + (size_t)e * I_DIM * ldW;
  const int*   uze = uz  + (size_t)e * I_DIM * ldG;
  const float* use = usc + (size_t)e * I_DIM * ldG;

  if (threadIdx.x < 16) toks[threadIdx.x] = rowmap[r0 + threadIdx.x];
  __syncthreads();

  v8f accg = {}; v8f accu = {};
  for (int kk = 0; kk < H_DIM; kk += 64) {
    { // gather-A: indirected token rows, pad rows -> zeros
#pragma unroll
      for (int rep = 0; rep < 2; ++rep) {
        int dw = rep * 256 + threadIdx.x;
        int m = dw >> 5;
        int kp = (dw & 31) * 2;
        int t = toks[m];
        uint32_t v = 0;
        if (t >= 0) v = *(const uint32_t*)(xh + (size_t)t * H_DIM + kk + kp);
        *(uint32_t*)(lds_a + m * 64 + kp) = v;
      }
    }
    dequant_tile64(gqe, gze, gse, ldW, ldG, iCol0, kk, lds_bg[wave], lane);
    dequant_tile64(uqe, uze, use, ldW, ldG, iCol0, kk, lds_bu[wave], lane);
    __syncthreads();
#pragma unroll
    for (int ko = 0; ko < 64; ko += 32) {
      v16h a = load_a_frag(lds_a, lane, ko);
      accg = wmma_f16(a, load_b_frag(lds_bg[wave], lane, ko), accg);
      accu = wmma_f16(a, load_b_frag(lds_bu[wave], lane, ko), accu);
    }
    __syncthreads();
  }
  const int n = lane & 15, mb = (lane >> 4) * 8;
#pragma unroll
  for (int r = 0; r < 8; ++r) {
    float hv = silu_f(accg[r]) * accu[r];
    hbuf[(size_t)(r0 + mb + r) * I_DIM + iCol0 + n] = (_Float16)hv;
  }
}

// ---------------- kernel 9: routed experts down-proj + weighted scatter ------
__global__ __launch_bounds__(256) void moe_down_kernel(
    const _Float16* __restrict__ hbuf, const int* __restrict__ rowmap,
    const float* __restrict__ roww, const int* __restrict__ tileExpert,
    const int* __restrict__ dq, const int* __restrict__ dz, const float* __restrict__ dsc,
    float* __restrict__ out)
{
  __shared__ _Float16 lds_a[16 * 64];
  __shared__ _Float16 lds_b[8][16 * 64];
  const int tile = blockIdx.x;
  const int e = tileExpert[tile];
  if (e < 0) return;
  const int wave = threadIdx.x >> 5, lane = threadIdx.x & 31;
  const int r0 = tile * 16;
  const int hCol0 = blockIdx.y * 128 + wave * 16;
  const int ldW = I_DIM / 8, ldG = I_DIM / GS_Q;   // 176, 11
  const int*   dqe = dq  + (size_t)e * H_DIM * ldW;
  const int*   dze = dz  + (size_t)e * H_DIM * ldG;
  const float* dse = dsc + (size_t)e * H_DIM * ldG;

  v8f acc = {};
  for (int kk = 0; kk < I_DIM; kk += 64) {
    load_a_tile(lds_a, hbuf + (size_t)r0 * I_DIM, I_DIM, kk, threadIdx.x);
    dequant_tile64(dqe, dze, dse, ldW, ldG, hCol0, kk, lds_b[wave], lane);
    __syncthreads();
#pragma unroll
    for (int ko = 0; ko < 64; ko += 32)
      acc = wmma_f16(load_a_frag(lds_a, lane, ko),
                     load_b_frag(lds_b[wave], lane, ko), acc);
    __syncthreads();
  }
  const int n = lane & 15, mb = (lane >> 4) * 8;
#pragma unroll
  for (int r = 0; r < 8; ++r) {
    int row = r0 + mb + r;
    int t = rowmap[row];
    if (t >= 0)
      atomicAdd(out + (size_t)t * H_DIM + hCol0 + n, roww[row] * acc[r]);
  }
}

// ---------------- host launch ----------------
extern "C" void kernel_launch(void* const* d_in, const int* in_sizes, int n_in,
                              void* d_out, int out_size, void* d_ws, size_t ws_size,
                              hipStream_t stream)
{
  (void)in_sizes; (void)n_in; (void)out_size; (void)ws_size;
  const float* x   = (const float*)d_in[0];
  const float* gw  = (const float*)d_in[1];
  const int*   gq  = (const int*)d_in[2];
  const int*   gz  = (const int*)d_in[3];
  const float* gsc = (const float*)d_in[4];
  const int*   uq  = (const int*)d_in[5];
  const int*   uz  = (const int*)d_in[6];
  const float* usc = (const float*)d_in[7];
  const int*   dq  = (const int*)d_in[8];
  const int*   dz  = (const int*)d_in[9];
  const float* dsc = (const float*)d_in[10];
  const float* sgw = (const float*)d_in[11];
  const float* suw = (const float*)d_in[12];
  const float* sdw = (const float*)d_in[13];
  float* out = (float*)d_out;

  uint8_t* ws = (uint8_t*)d_ws;
  size_t off = 0;
  auto alloc = [&](size_t bytes) -> void* {
    void* p = (void*)(ws + off);
    off += (bytes + 255) & ~(size_t)255;
    return p;
  };
  int*      counts     = (int*)alloc(E_NUM * sizeof(int));
  int*      offsets    = (int*)alloc((E_NUM + 1) * sizeof(int));
  int*      cursor     = (int*)alloc(E_NUM * sizeof(int));
  int*      tileExpert = (int*)alloc(MAX_TILES * sizeof(int));
  int*      rowmap     = (int*)alloc(MAX_ROWS * sizeof(int));
  float*    roww       = (float*)alloc(MAX_ROWS * sizeof(float));
  int*      tokexp     = (int*)alloc(T_TOK * TOPK * sizeof(int));
  float*    tokw       = (float*)alloc(T_TOK * TOPK * sizeof(float));
  _Float16* xh         = (_Float16*)alloc((size_t)T_TOK * H_DIM * 2);
  _Float16* hbuf       = (_Float16*)alloc((size_t)MAX_ROWS * I_DIM * 2);
  _Float16* hsbuf      = (_Float16*)alloc((size_t)T_TOK * ISH_DIM * 2);
  (void)offsets;

  zero_counts_kernel<<<1, 32, 0, stream>>>(counts);
  router_kernel<<<T_TOK, 32, 0, stream>>>(x, gw, tokexp, tokw, counts);
  sched_kernel<<<1, 256, 0, stream>>>(counts, offsets, cursor, tileExpert, rowmap, roww);
  fill_rows_kernel<<<(T_TOK * TOPK + 255) / 256, 256, 0, stream>>>(
      tokexp, tokw, cursor, rowmap, roww);
  cvt_x_kernel<<<(T_TOK * H_DIM + 255) / 256, 256, 0, stream>>>(x, xh);

  shared_gateup_kernel<<<dim3(T_TOK / 16, ISH_DIM / 128), 256, 0, stream>>>(
      xh, sgw, suw, hsbuf);
  shared_down_kernel<<<dim3(T_TOK / 16, H_DIM / 128), 256, 0, stream>>>(
      hsbuf, sdw, out);

  moe_gateup_kernel<<<dim3(MAX_TILES, I_DIM / 128), 256, 0, stream>>>(
      xh, rowmap, tileExpert, gq, gz, gsc, uq, uz, usc, hbuf);
  moe_down_kernel<<<dim3(MAX_TILES, H_DIM / 128), 256, 0, stream>>>(
      hbuf, rowmap, roww, tileExpert, dq, dz, dsc, out);
}
